// TransformerBlock_39067022524456
// MI455X (gfx1250) — compile-verified, matrix operand order checked by reading, not run on hardware
//
#include <hip/hip_runtime.h>

// ---------------- constants -----------------
constexpr int BB = 4;
constexpr int TT = 2048;
constexpr int CC = 1024;
constexpr int FF = 4096;

// ---------------- types ---------------------
typedef __attribute__((ext_vector_type(16))) __bf16 v16bf;
typedef __attribute__((ext_vector_type(8)))  __bf16 v8bf;
typedef __attribute__((ext_vector_type(8)))  float  v8f;

// ---------------- helpers -------------------
__device__ __forceinline__ unsigned short f2bf(float f) {
  unsigned int u = __float_as_uint(f);
  u += 0x7FFFu + ((u >> 16) & 1u);
  return (unsigned short)(u >> 16);
}

__device__ __forceinline__ v8f wmma_bf16(v16bf a, v16bf b, v8f c) {
  return __builtin_amdgcn_wmma_f32_16x16x32_bf16(false, a, false, b, (short)0, c,
                                                 false, false);
}

// Load a 16x32 bf16 operand tile (A layout; also B layout of the transpose).
// p points at element [0][0] of the tile in a row-major matrix of row length
// `stride` (elements). lane&15 selects the row (A) / column-of-B (Bt row).
__device__ __forceinline__ v16bf ld16x32(const unsigned short* p, int stride, int lane) {
  const int r = lane & 15;
  const int g = (lane >> 4) * 8;
  const unsigned short* q = p + (size_t)r * stride + g;
  v8bf lo = *(const v8bf*)(q);
  v8bf hi = *(const v8bf*)(q + 16);
  v16bf out;
#pragma unroll
  for (int i = 0; i < 8; i++) { out[i] = lo[i]; out[8 + i] = hi[i]; }
  return out;
}

// ---------------- LN1 + RoPE ----------------
__global__ __launch_bounds__(256) void ln_rope_kernel(
    const float* __restrict__ x, const float* __restrict__ g,
    const float* __restrict__ be, unsigned short* __restrict__ Qr,
    unsigned short* __restrict__ Vv) {
  __shared__ float red[18];
  __shared__ float yy[CC];
  const int row = blockIdx.x;        // b*T + t
  const int tid = threadIdx.x;
  const int t = row & (TT - 1);
  const float* xr = x + (size_t)row * CC;

  float v[4], s0 = 0.f, s1 = 0.f;
#pragma unroll
  for (int j = 0; j < 4; j++) {
    v[j] = xr[tid + 256 * j];
    s0 += v[j]; s1 += v[j] * v[j];
  }
#pragma unroll
  for (int off = 16; off; off >>= 1) {
    s0 += __shfl_xor(s0, off, 32);
    s1 += __shfl_xor(s1, off, 32);
  }
  if ((tid & 31) == 0) { red[tid >> 5] = s0; red[8 + (tid >> 5)] = s1; }
  __syncthreads();
  if (tid == 0) {
    float a = 0.f, q = 0.f;
    for (int i = 0; i < 8; i++) { a += red[i]; q += red[8 + i]; }
    float mu = a * (1.0f / CC);
    float var = q * (1.0f / CC) - mu * mu;
    red[16] = mu; red[17] = rsqrtf(var + 1e-5f);
  }
  __syncthreads();
  const float mu = red[16], rs = red[17];
#pragma unroll
  for (int j = 0; j < 4; j++) {
    int c = tid + 256 * j;
    float y = (v[j] - mu) * rs * g[c] + be[c];
    yy[c] = y;
    Vv[(size_t)row * CC + c] = f2bf(y);
  }
  __syncthreads();
#pragma unroll
  for (int j = 0; j < 4; j++) {
    int c = tid + 256 * j;
    int i = c & 511;
    float inv = __expf(-(float)i * (9.210340371976184f / 512.0f));
    float ang = (float)t * inv;
    float sn, cs;
    __sincosf(ang, &sn, &cs);
    float rot = (c < 512) ? -yy[c + 512] : yy[c - 512];
    Qr[(size_t)row * CC + c] = f2bf(yy[c] * cs + rot * sn);
  }
}

// ---------------- LN2 -----------------------
__global__ __launch_bounds__(256) void ln_kernel(
    const float* __restrict__ in, const float* __restrict__ g,
    const float* __restrict__ be, unsigned short* __restrict__ out) {
  __shared__ float red[18];
  const int row = blockIdx.x;
  const int tid = threadIdx.x;
  const float* xr = in + (size_t)row * CC;
  float v[4], s0 = 0.f, s1 = 0.f;
#pragma unroll
  for (int j = 0; j < 4; j++) {
    v[j] = xr[tid + 256 * j];
    s0 += v[j]; s1 += v[j] * v[j];
  }
#pragma unroll
  for (int off = 16; off; off >>= 1) {
    s0 += __shfl_xor(s0, off, 32);
    s1 += __shfl_xor(s1, off, 32);
  }
  if ((tid & 31) == 0) { red[tid >> 5] = s0; red[8 + (tid >> 5)] = s1; }
  __syncthreads();
  if (tid == 0) {
    float a = 0.f, q = 0.f;
    for (int i = 0; i < 8; i++) { a += red[i]; q += red[8 + i]; }
    float mu = a * (1.0f / CC);
    float var = q * (1.0f / CC) - mu * mu;
    red[16] = mu; red[17] = rsqrtf(var + 1e-5f);
  }
  __syncthreads();
  const float mu = red[16], rs = red[17];
#pragma unroll
  for (int j = 0; j < 4; j++) {
    int c = tid + 256 * j;
    out[(size_t)row * CC + c] = f2bf((v[j] - mu) * rs * g[c] + be[c]);
  }
}

// ------------- transposes -------------------
// bf16 [rows,cols] -> bf16 [cols,rows] (per z-batch of rows*cols)
__global__ __launch_bounds__(256) void tbf16_kernel(
    const unsigned short* __restrict__ in, unsigned short* __restrict__ out,
    int rows, int cols) {
  __shared__ unsigned short tile[32][33];
  const int r0 = blockIdx.y * 32, c0 = blockIdx.x * 32;
  const size_t base = (size_t)blockIdx.z * rows * cols;
  const int tx = threadIdx.x & 31, ty = threadIdx.x >> 5;
#pragma unroll
  for (int k = 0; k < 4; k++) {
    int r = ty + 8 * k;
    tile[r][tx] = in[base + (size_t)(r0 + r) * cols + c0 + tx];
  }
  __syncthreads();
#pragma unroll
  for (int k = 0; k < 4; k++) {
    int r = ty + 8 * k;
    out[base + (size_t)(c0 + r) * rows + r0 + tx] = tile[tx][r];
  }
}

// f32 [rows,cols] -> bf16 [cols,rows]
__global__ __launch_bounds__(256) void tf32bf_kernel(
    const float* __restrict__ in, unsigned short* __restrict__ out,
    int rows, int cols) {
  __shared__ unsigned short tile[32][33];
  const int r0 = blockIdx.y * 32, c0 = blockIdx.x * 32;
  const int tx = threadIdx.x & 31, ty = threadIdx.x >> 5;
#pragma unroll
  for (int k = 0; k < 4; k++) {
    int r = ty + 8 * k;
    tile[r][tx] = f2bf(in[(size_t)(r0 + r) * cols + c0 + tx]);
  }
  __syncthreads();
#pragma unroll
  for (int k = 0; k < 4; k++) {
    int r = ty + 8 * k;
    out[(size_t)(c0 + r) * rows + r0 + tx] = tile[tx][r];
  }
}

// ------------- flash attention --------------
// Grid: (T/32, B). Block: 256 (8 waves). Wave w: S tile (mt=w>>2, nt=w&3),
// O slice = all 32 rows x columns [w*128, w*128+128).
__global__ __launch_bounds__(256) void attn_kernel(
    const unsigned short* __restrict__ Qr,   // [B,T,C] bf16 (q == k)
    const unsigned short* __restrict__ Vt,   // [B,C,T] bf16 (V transposed)
    const float* __restrict__ x,             // residual
    float* __restrict__ xo) {                // x + attn out
  __shared__ __align__(16) unsigned short sP[32 * 64];
  __shared__ float pmax[4][32];
  __shared__ float psum[4][32];
  __shared__ float rowM[32], rowL[32], rowC[32];

  const int b = blockIdx.y;
  const int q0 = blockIdx.x * 32;
  const int tid = threadIdx.x, lane = tid & 31, w = tid >> 5;
  const int mt = w >> 2, nt = w & 3;
  const int hlf = lane >> 4, nn = lane & 15;

  const unsigned short* Qb = Qr + (size_t)b * TT * CC;
  const unsigned short* Vtb = Vt + (size_t)b * CC * TT;

  if (tid < 32) { rowM[tid] = -3.0e38f; rowL[tid] = 0.f; }
  v8f o[2][8];
#pragma unroll
  for (int mi = 0; mi < 2; mi++)
#pragma unroll
    for (int ni = 0; ni < 8; ni++)
#pragma unroll
      for (int r = 0; r < 8; r++) o[mi][ni][r] = 0.f;
  __syncthreads();

  const unsigned short* Ab = Qb + (size_t)(q0 + mt * 16) * CC;

  for (int kb = 0; kb < TT / 64; kb++) {
    const int krow = kb * 64;
    const unsigned short* Bb = Qb + (size_t)(krow + nt * 16) * CC;
    if (kb + 1 < TT / 64) {  // CDNA5 global_prefetch of next key block
      __builtin_prefetch(Qb + (size_t)(krow + 64 + nt * 16) * CC, 0, 1);
      __builtin_prefetch(Vtb + (size_t)(w * 128) * TT + krow + 64, 0, 1);
    }

    // ---- S = Q . K^T (K reduction over C), software pipelined ----
    v8f s;
#pragma unroll
    for (int r = 0; r < 8; r++) s[r] = 0.f;
    v16bf a_c = ld16x32(Ab, CC, lane);
    v16bf b_c = ld16x32(Bb, CC, lane);
#pragma unroll 4
    for (int k0 = 0; k0 < CC; k0 += 32) {
      v16bf a_n = a_c, b_n = b_c;
      if (k0 + 32 < CC) {
        a_n = ld16x32(Ab + k0 + 32, CC, lane);
        b_n = ld16x32(Bb + k0 + 32, CC, lane);
      }
      s = wmma_bf16(a_c, b_c, s);
      a_c = a_n; b_c = b_n;
    }

    // ---- scale + per-tile row max ----
    float sm[8], tmax8[8];
#pragma unroll
    for (int r = 0; r < 8; r++) {
      sm[r] = s[r] * 0.03125f;  // 1/sqrt(1024)
      float v = sm[r];
      v = fmaxf(v, __shfl_xor(v, 1, 32));
      v = fmaxf(v, __shfl_xor(v, 2, 32));
      v = fmaxf(v, __shfl_xor(v, 4, 32));
      v = fmaxf(v, __shfl_xor(v, 8, 32));
      tmax8[r] = v;
    }
    if (nn == 0) {
#pragma unroll
      for (int r = 0; r < 8; r++) pmax[nt][mt * 16 + hlf * 8 + r] = tmax8[r];
    }
    __syncthreads();
    if (tid < 32) {
      float bm = fmaxf(fmaxf(pmax[0][tid], pmax[1][tid]),
                       fmaxf(pmax[2][tid], pmax[3][tid]));
      float mo = rowM[tid];
      float mn = fmaxf(mo, bm);
      rowC[tid] = __expf(mo - mn);
      rowM[tid] = mn;
    }
    __syncthreads();

    // ---- P = exp(S - m), stage bf16 P in LDS, partial row sums ----
    float rsum[8];
#pragma unroll
    for (int r = 0; r < 8; r++) {
      int mr = mt * 16 + hlf * 8 + r;
      float p = __expf(sm[r] - rowM[mr]);
      sP[mr * 64 + nt * 16 + nn] = f2bf(p);
      float tsum = p;
      tsum += __shfl_xor(tsum, 1, 32);
      tsum += __shfl_xor(tsum, 2, 32);
      tsum += __shfl_xor(tsum, 4, 32);
      tsum += __shfl_xor(tsum, 8, 32);
      rsum[r] = tsum;
    }
    if (nn == 0) {
#pragma unroll
      for (int r = 0; r < 8; r++) psum[nt][mt * 16 + hlf * 8 + r] = rsum[r];
    }
    __syncthreads();
    if (tid < 32) {
      rowL[tid] = rowL[tid] * rowC[tid] + psum[0][tid] + psum[1][tid] +
                  psum[2][tid] + psum[3][tid];
    }

    // ---- rescale O by exp(m_old - m_new) ----
#pragma unroll
    for (int mi = 0; mi < 2; mi++) {
#pragma unroll
      for (int r = 0; r < 8; r++) {
        float cf = rowC[mi * 16 + hlf * 8 + r];
#pragma unroll
        for (int ni = 0; ni < 8; ni++) o[mi][ni][r] *= cf;
      }
    }

    // ---- O += P . V  (B operand from Vt), pipelined over 16 tiles ----
    v16bf pa[2][2];
#pragma unroll
    for (int mi = 0; mi < 2; mi++)
#pragma unroll
      for (int ks = 0; ks < 2; ks++)
        pa[mi][ks] = ld16x32(sP + (mi * 16) * 64 + ks * 32, 64, lane);

    const unsigned short* vbase = Vtb + (size_t)(w * 128) * TT + krow;
    v16bf vb = ld16x32(vbase, TT, lane);  // u=0: ni=0, ks=0
#pragma unroll
    for (int u = 0; u < 16; u++) {
      const int ni = u >> 1, ks = u & 1;
      v16bf nb = vb;
      if (u < 15) {
        const int ni2 = (u + 1) >> 1, ks2 = (u + 1) & 1;
        nb = ld16x32(vbase + (size_t)(ni2 * 16) * TT + ks2 * 32, TT, lane);
      }
      o[0][ni] = wmma_bf16(pa[0][ks], vb, o[0][ni]);
      o[1][ni] = wmma_bf16(pa[1][ks], vb, o[1][ni]);
      vb = nb;
    }
    __syncthreads();
  }

  // ---- epilogue: O/l + residual ----
  const float* xr = x + (size_t)(b * TT + q0) * CC;
  float* xw = xo + (size_t)(b * TT + q0) * CC;
#pragma unroll
  for (int mi = 0; mi < 2; mi++) {
#pragma unroll
    for (int r = 0; r < 8; r++) {
      int mr = mi * 16 + hlf * 8 + r;
      float inv = 1.0f / rowL[mr];
#pragma unroll
      for (int ni = 0; ni < 8; ni++) {
        int col = w * 128 + ni * 16 + nn;
        xw[(size_t)mr * CC + col] = xr[(size_t)mr * CC + col] + o[mi][ni][r] * inv;
      }
    }
  }
}

// ------------- WMMA GEMM --------------------
// D = A[M,K] @ Bt[N,K]^T.  epi==0: out_bf16 = GELU(D + bias)
//                          epi==1: out_f32  = D + bias + resid
// Block tile 128x128, 8 waves as 2(M) x 4(N), wave tile 64x32, K step 32.
// Double-buffered LDS: next tile's global loads overlap current WMMAs.
__global__ __launch_bounds__(256) void gemm_kernel(
    const unsigned short* __restrict__ A, const unsigned short* __restrict__ Bt,
    const float* __restrict__ bias, const float* __restrict__ resid,
    unsigned short* __restrict__ outb, float* __restrict__ outf,
    int M, int N, int K, int epi) {
  __shared__ __align__(16) unsigned short sA[2][128 * 32];
  __shared__ __align__(16) unsigned short sB[2][128 * 32];
  const int m0 = blockIdx.y * 128, n0 = blockIdx.x * 128;
  const int tid = threadIdx.x, lane = tid & 31, w = tid >> 5;
  const int wm = w >> 2, wn = w & 3;
  const int hlf = lane >> 4, nn = lane & 15;

  v8f acc[4][2];
#pragma unroll
  for (int mi = 0; mi < 4; mi++)
#pragma unroll
    for (int ni = 0; ni < 2; ni++)
#pragma unroll
      for (int r = 0; r < 8; r++) acc[mi][ni][r] = 0.f;

  const int r0 = (tid + 0) >> 2,   s0 = (tid + 0) & 3;
  const int r1 = (tid + 256) >> 2, s1 = (tid + 256) & 3;
  const int nsteps = K >> 5;

  uint4 ra0, ra1, rb0, rb1;
  // prologue: stage k-step 0
  ra0 = *(const uint4*)(A  + (size_t)(m0 + r0) * K + s0 * 8);
  ra1 = *(const uint4*)(A  + (size_t)(m0 + r1) * K + s1 * 8);
  rb0 = *(const uint4*)(Bt + (size_t)(n0 + r0) * K + s0 * 8);
  rb1 = *(const uint4*)(Bt + (size_t)(n0 + r1) * K + s1 * 8);
  *(uint4*)(sA[0] + r0 * 32 + s0 * 8) = ra0;
  *(uint4*)(sA[0] + r1 * 32 + s1 * 8) = ra1;
  *(uint4*)(sB[0] + r0 * 32 + s0 * 8) = rb0;
  *(uint4*)(sB[0] + r1 * 32 + s1 * 8) = rb1;
  __syncthreads();

  for (int ks = 0; ks < nsteps; ks++) {
    const int cur = ks & 1;
    // issue next tile's global loads before computing
    if (ks + 1 < nsteps) {
      const int k0n = (ks + 1) << 5;
      ra0 = *(const uint4*)(A  + (size_t)(m0 + r0) * K + k0n + s0 * 8);
      ra1 = *(const uint4*)(A  + (size_t)(m0 + r1) * K + k0n + s1 * 8);
      rb0 = *(const uint4*)(Bt + (size_t)(n0 + r0) * K + k0n + s0 * 8);
      rb1 = *(const uint4*)(Bt + (size_t)(n0 + r1) * K + k0n + s1 * 8);
    }
    // compute from current LDS buffer
    v16bf aa[4], bb[2];
#pragma unroll
    for (int mi = 0; mi < 4; mi++)
      aa[mi] = ld16x32(sA[cur] + (wm * 64 + mi * 16) * 32, 32, lane);
#pragma unroll
    for (int ni = 0; ni < 2; ni++)
      bb[ni] = ld16x32(sB[cur] + (wn * 32 + ni * 16) * 32, 32, lane);
#pragma unroll
    for (int mi = 0; mi < 4; mi++)
#pragma unroll
      for (int ni = 0; ni < 2; ni++)
        acc[mi][ni] = wmma_bf16(aa[mi], bb[ni], acc[mi][ni]);
    // drain staged registers into the alternate buffer
    if (ks + 1 < nsteps) {
      const int nxt = cur ^ 1;
      *(uint4*)(sA[nxt] + r0 * 32 + s0 * 8) = ra0;
      *(uint4*)(sA[nxt] + r1 * 32 + s1 * 8) = ra1;
      *(uint4*)(sB[nxt] + r0 * 32 + s0 * 8) = rb0;
      *(uint4*)(sB[nxt] + r1 * 32 + s1 * 8) = rb1;
    }
    __syncthreads();
  }

#pragma unroll
  for (int mi = 0; mi < 4; mi++) {
#pragma unroll
    for (int ni = 0; ni < 2; ni++) {
      int col = n0 + wn * 32 + ni * 16 + nn;
      float bv = bias[col];
#pragma unroll
      for (int r = 0; r < 8; r++) {
        int row = m0 + wm * 64 + mi * 16 + hlf * 8 + r;
        float v = acc[mi][ni][r] + bv;
        if (epi == 0) {
          float gv = 0.5f * v * (1.0f + erff(v * 0.70710678118654752f));
          outb[(size_t)row * N + col] = f2bf(gv);
        } else {
          outf[(size_t)row * N + col] = v + resid[(size_t)row * N + col];
        }
      }
    }
  }
}

// ---------------- launcher ------------------
extern "C" void kernel_launch(void* const* d_in, const int* in_sizes, int n_in,
                              void* d_out, int out_size, void* d_ws, size_t ws_size,
                              hipStream_t stream) {
  (void)in_sizes; (void)n_in; (void)out_size; (void)ws_size;
  const float* x    = (const float*)d_in[0];
  const float* ln1g = (const float*)d_in[1];
  const float* ln1b = (const float*)d_in[2];
  const float* ln2g = (const float*)d_in[3];
  const float* ln2b = (const float*)d_in[4];
  const float* W1   = (const float*)d_in[5];
  const float* b1   = (const float*)d_in[6];
  const float* W2   = (const float*)d_in[7];
  const float* b2   = (const float*)d_in[8];
  float* out = (float*)d_out;

  char* ws = (char*)d_ws;
  const size_t sBTC_bf = (size_t)BB * TT * CC * 2;   // 16 MB
  unsigned short* Qr  = (unsigned short*)(ws);                       // [B,T,C]
  unsigned short* Vv  = (unsigned short*)(ws + sBTC_bf);             // [B,T,C]
  unsigned short* Vt  = (unsigned short*)(ws + 2 * sBTC_bf);         // [B,C,T]
  float*          Xa  = (float*)(ws + 3 * sBTC_bf);                  // [B,T,C] f32
  unsigned short* H   = (unsigned short*)(ws + 3 * sBTC_bf + (size_t)BB * TT * CC * 4);
  unsigned short* W1t = (unsigned short*)((char*)H + sBTC_bf);       // [F,C]
  unsigned short* W2t = (unsigned short*)((char*)W1t + (size_t)CC * FF * 2); // [C,F]
  unsigned short* Act = (unsigned short*)((char*)W2t + (size_t)CC * FF * 2); // [B*T,F]

  // 1) LN1 + RoPE -> Q(=K), V in bf16
  ln_rope_kernel<<<BB * TT, 256, 0, stream>>>(x, ln1g, ln1b, Qr, Vv);
  // 2) V -> V^T per batch
  tbf16_kernel<<<dim3(CC / 32, TT / 32, BB), 256, 0, stream>>>(Vv, Vt, TT, CC);
  // 3) weights -> bf16 transposed
  tf32bf_kernel<<<dim3(FF / 32, CC / 32, 1), 256, 0, stream>>>(W1, W1t, CC, FF);
  tf32bf_kernel<<<dim3(CC / 32, FF / 32, 1), 256, 0, stream>>>(W2, W2t, FF, CC);
  // 4) flash attention + residual -> Xa
  attn_kernel<<<dim3(TT / 32, BB), 256, 0, stream>>>(Qr, Vt, x, Xa);
  // 5) LN2 -> H bf16
  ln_kernel<<<BB * TT, 256, 0, stream>>>(Xa, ln2g, ln2b, H);
  // 6) GEMM1 + GELU -> Act bf16   (M=8192, N=4096, K=1024)
  gemm_kernel<<<dim3(FF / 128, (BB * TT) / 128), 256, 0, stream>>>(
      H, W1t, b1, nullptr, Act, nullptr, BB * TT, FF, CC, 0);
  // 7) GEMM2 + bias + residual -> out f32   (M=8192, N=1024, K=4096)
  gemm_kernel<<<dim3(CC / 128, (BB * TT) / 128), 256, 0, stream>>>(
      Act, W2t, b2, Xa, nullptr, out, BB * TT, CC, FF, 1);
}
